// FusedLinearCrossEntropy_72224170050156
// MI455X (gfx1250) — matrix operand order repaired and checked by reading, not compiled, not run on hardware
//
#include <hip/hip_runtime.h>
#include <hip/hip_bf16.h>
#include <math.h>

// ---------------------------------------------------------------------------
// Fused Linear + CrossEntropy (+ z-loss) for MI455X (gfx1250, wave32, WMMA)
//
// logits = input[N,D] @ weight[V,D]^T + bias[V]
// loss   = mean_valid( lse - tgt_logit + 1e-4*lse^2 ), z = mean_valid(1e-4*lse^2)
//
// LDS-double-tiled GEMM on v_wmma_f32_16x16x32_bf16:
//   block(512 thr/16 waves) computes a 128x256 logits tile; per 64-wide K-slab
//   A is staged to LDS as bf16 hi+lo (split precision), B as bf16 hi, both in
//   pre-swizzled WMMA fragment order (conflict-free ds_load_b128 frag loads).
//   Per-row online-softmax partials (m, s) are written per 256-col chunk and
//   merged by a second kernel. Target logits come from a cheap dot kernel.
// ---------------------------------------------------------------------------

typedef __attribute__((ext_vector_type(16))) __bf16 v16bf;
typedef __attribute__((ext_vector_type(8)))  __bf16 v8bf;
typedef __attribute__((ext_vector_type(8)))  float  v8f;

#define IGNORE_INDEX (-100)
#define Z_SCALE 1e-4f
#define D_DIM 2048
#define BLK_ROWS 128
#define BLK_COLS 256
#define KSLAB 64
#define NTHREADS 512
#define NEG_INF (-__builtin_inff())

__device__ __forceinline__ float redmax16(float v) {
  v = fmaxf(v, __shfl_xor(v, 1, 32));
  v = fmaxf(v, __shfl_xor(v, 2, 32));
  v = fmaxf(v, __shfl_xor(v, 4, 32));
  v = fmaxf(v, __shfl_xor(v, 8, 32));
  return v;
}
__device__ __forceinline__ float redsum16(float v) {
  v += __shfl_xor(v, 1, 32);
  v += __shfl_xor(v, 2, 32);
  v += __shfl_xor(v, 4, 32);
  v += __shfl_xor(v, 8, 32);
  return v;
}
__device__ __forceinline__ float redsum32(float v) {
  v += __shfl_xor(v, 1, 32);
  v += __shfl_xor(v, 2, 32);
  v += __shfl_xor(v, 4, 32);
  v += __shfl_xor(v, 8, 32);
  v += __shfl_xor(v, 16, 32);
  return v;
}

__device__ __forceinline__ v16bf mk16(v8bf a, v8bf b) {
  v16bf r;
#pragma unroll
  for (int i = 0; i < 8; ++i) { r[i] = a[i]; r[i + 8] = b[i]; }
  return r;
}
__device__ __forceinline__ v8bf cvt_hi8(const float* f) {
  v8bf r;
#pragma unroll
  for (int i = 0; i < 8; ++i) r[i] = (__bf16)f[i];
  return r;
}
__device__ __forceinline__ void cvt_hilo8(const float* f, v8bf& h8, v8bf& l8) {
#pragma unroll
  for (int i = 0; i < 8; ++i) {
    __bf16 h = (__bf16)f[i];
    h8[i] = h;
    l8[i] = (__bf16)(f[i] - (float)h);
  }
}

// -------------------- init / finalize --------------------

__global__ void flce_init_kernel(float* acc) {
  if (threadIdx.x < 4) acc[threadIdx.x] = 0.0f;
}

__global__ void flce_finalize_kernel(const float* acc, float* out) {
  if (threadIdx.x == 0) {
    float n = fmaxf(acc[2], 1.0f);
    out[0] = acc[0] / n;  // loss (ce + z-loss mean)
    out[1] = acc[1] / n;  // z_loss mean
  }
}

// -------------------- target logit: one wave per row --------------------

__global__ void flce_tgt_kernel(const float* __restrict__ input,
                                const float* __restrict__ weight,
                                const float* __restrict__ bias,
                                const int* __restrict__ target,
                                float* __restrict__ tgtl, int N, int V) {
  int wave = threadIdx.x >> 5;
  int lane = threadIdx.x & 31;
  int row = blockIdx.x * 8 + wave;
  if (row >= N) return;
  int t = target[row];
  if (t < 0 || t >= V) {
    if (lane == 0) tgtl[row] = 0.0f;
    return;
  }
  const float4* ip = (const float4*)(input + (size_t)row * D_DIM);
  const float4* wp = (const float4*)(weight + (size_t)t * D_DIM);
  float s = 0.0f;
  for (int k4 = lane; k4 < D_DIM / 4; k4 += 32) {
    float4 a = ip[k4], b = wp[k4];
    s += a.x * b.x + a.y * b.y + a.z * b.z + a.w * b.w;
  }
  s = redsum32(s);
  if (lane == 0) tgtl[row] = s + bias[t];
}

// -------------------- main fused GEMM + LSE partials --------------------
// grid = (N/128, ceil(V/256)), block = 512 (16 waves).
// Wave w: row-group rg = w&3 (32 rows), col-group cg = w>>2 (64 cols).
// LDS fragment layout (v8bf units): idx(tile, ks, h) = (tile*4 + ks*2 + h)*32 + lane
// A-frag lane map: lane = (r&15) + 16*((kk>>3)&1), h = (kk&31)>>4, e = kk&7
// B-frag lane map: lane = (c&15) + 16*((kk>>3)&1), h = (kk>>4)&1,  e = kk&7

__global__ __launch_bounds__(NTHREADS)
void flce_main_kernel(const float* __restrict__ input,
                      const float* __restrict__ weight,
                      const float* __restrict__ bias,
                      float* __restrict__ pm_out, float* __restrict__ ps_out,
                      int V, int nchunk) {
  __shared__ v8bf AhiV[8 * 4 * 32];   // 8 row-tiles x 2 ks x 2 h x 32 lanes (16KB)
  __shared__ v8bf AloV[8 * 4 * 32];   // 16KB
  __shared__ v8bf BhV[16 * 4 * 32];   // 16 col-tiles (32KB)
  __shared__ float Pm[4][BLK_ROWS];
  __shared__ float Ps[4][BLK_ROWS];

  const int tid = threadIdx.x;
  const int wave = tid >> 5;
  const int lane = tid & 31;
  const int rg = wave & 3;   // row group: rows rg*32 .. rg*32+31
  const int cg = wave >> 2;  // col group: cols cg*64 .. cg*64+63
  const int rbase = blockIdx.x * BLK_ROWS;
  const int vbase = blockIdx.y * BLK_COLS;

  v8f acc[2][4];
#pragma unroll
  for (int i = 0; i < 2; ++i)
#pragma unroll
    for (int j = 0; j < 4; ++j) acc[i][j] = {};

  // staging coordinates (computed once)
  const int ar = tid >> 2;               // A row 0..127
  const int akh = (tid & 3) << 4;        // A k-offset {0,16,32,48}
  const int bc = tid >> 1;               // B col 0..255
  const int bkh = (tid & 1) << 5;        // B k-offset {0,32}
  const int bwrow = (vbase + bc < V) ? (vbase + bc) : (V - 1);

#pragma unroll 1
  for (int slab = 0; slab < D_DIM / KSLAB; ++slab) {
    const int k0 = slab * KSLAB;
    __syncthreads();  // previous slab fully consumed

    // ---- stage A (hi + lo) ----
    {
      float fl[16];
      const float4* p =
          (const float4*)(input + (size_t)(rbase + ar) * D_DIM + k0 + akh);
#pragma unroll
      for (int j = 0; j < 4; ++j) {
        float4 q = p[j];
        fl[4 * j] = q.x; fl[4 * j + 1] = q.y;
        fl[4 * j + 2] = q.z; fl[4 * j + 3] = q.w;
      }
#pragma unroll
      for (int g4 = 0; g4 < 2; ++g4) {
        int kk = akh + g4 * 8;
        int ks = kk >> 5;
        int g = (kk & 31) >> 3;  // 0..3
        int vidx = (((ar >> 4) * 2 + ks) * 2 + (g >> 1)) * 32 +
                   (ar & 15) + ((g & 1) << 4);
        v8bf h8, l8;
        cvt_hilo8(&fl[g4 * 8], h8, l8);
        AhiV[vidx] = h8;
        AloV[vidx] = l8;
      }
    }
    // ---- stage B (hi only) ----
    {
      float fl[32];
      const float4* p =
          (const float4*)(weight + (size_t)bwrow * D_DIM + k0 + bkh);
#pragma unroll
      for (int j = 0; j < 8; ++j) {
        float4 q = p[j];
        fl[4 * j] = q.x; fl[4 * j + 1] = q.y;
        fl[4 * j + 2] = q.z; fl[4 * j + 3] = q.w;
      }
#pragma unroll
      for (int g8 = 0; g8 < 4; ++g8) {
        int kk = bkh + g8 * 8;
        int vidx = (((bc >> 4) * 2 + (kk >> 5)) * 2 + ((kk >> 4) & 1)) * 32 +
                   (bc & 15) + (((kk >> 3) & 1) << 4);
        BhV[vidx] = cvt_hi8(&fl[g8 * 8]);
      }
    }
    __syncthreads();

    // ---- compute: 2 ks x 4 col-tiles x (hi + lo A) x 2 row-tiles = 32 WMMA
#pragma unroll
    for (int ks = 0; ks < 2; ++ks) {
      const int a0 = ((rg * 2 + 0) * 4 + ks * 2) * 32 + lane;
      const int a1 = ((rg * 2 + 1) * 4 + ks * 2) * 32 + lane;
      v16bf ah0 = mk16(AhiV[a0], AhiV[a0 + 32]);
      v16bf al0 = mk16(AloV[a0], AloV[a0 + 32]);
      v16bf ah1 = mk16(AhiV[a1], AhiV[a1 + 32]);
      v16bf al1 = mk16(AloV[a1], AloV[a1 + 32]);
#pragma unroll
      for (int ctl = 0; ctl < 4; ++ctl) {
        const int bb = ((cg * 4 + ctl) * 4 + ks * 2) * 32 + lane;
        v16bf b = mk16(BhV[bb], BhV[bb + 32]);
        acc[0][ctl] = __builtin_amdgcn_wmma_f32_16x16x32_bf16(
            false, ah0, false, b, (short)0, acc[0][ctl], false, false);
        acc[0][ctl] = __builtin_amdgcn_wmma_f32_16x16x32_bf16(
            false, al0, false, b, (short)0, acc[0][ctl], false, false);
        acc[1][ctl] = __builtin_amdgcn_wmma_f32_16x16x32_bf16(
            false, ah1, false, b, (short)0, acc[1][ctl], false, false);
        acc[1][ctl] = __builtin_amdgcn_wmma_f32_16x16x32_bf16(
            false, al1, false, b, (short)0, acc[1][ctl], false, false);
      }
    }
  }

  // ---- epilogue: masked row max / sum-exp over this wave's 64 cols ----
  // C layout: VGPR r, lane L -> row = rowtile*16 + r + 8*(L>=16), col = (L&15)
  const int lh = lane & 15;
  bool okc[4];
  float bc4[4];
#pragma unroll
  for (int ctl = 0; ctl < 4; ++ctl) {
    int col = vbase + cg * 64 + ctl * 16 + lh;
    okc[ctl] = col < V;
    bc4[ctl] = bias[okc[ctl] ? col : (V - 1)];
  }

#pragma unroll
  for (int rtl = 0; rtl < 2; ++rtl) {
#pragma unroll
    for (int r = 0; r < 8; ++r) {
      float x[4];
#pragma unroll
      for (int ctl = 0; ctl < 4; ++ctl)
        x[ctl] = okc[ctl] ? (acc[rtl][ctl][r] + bc4[ctl]) : NEG_INF;
      float m = fmaxf(fmaxf(x[0], x[1]), fmaxf(x[2], x[3]));
      m = redmax16(m);
      float pe = 0.0f;
#pragma unroll
      for (int ctl = 0; ctl < 4; ++ctl)
        pe += okc[ctl] ? expf(x[ctl] - m) : 0.0f;  // exact 0 for masked cols
      pe = redsum16(pe);
      int rowloc = rg * 32 + rtl * 16 + r;
      if (lane == 0) { Pm[cg][rowloc] = m; Ps[cg][rowloc] = pe; }
      else if (lane == 16) { Pm[cg][rowloc + 8] = m; Ps[cg][rowloc + 8] = pe; }
    }
  }
  __syncthreads();

  // ---- merge 4 col-groups, emit per-(row, chunk) partials ----
  if (tid < BLK_ROWS) {
    int row = tid;
    float M = NEG_INF;
#pragma unroll
    for (int g = 0; g < 4; ++g) M = fmaxf(M, Pm[g][row]);
    float S = 0.0f;
#pragma unroll
    for (int g = 0; g < 4; ++g) S += Ps[g][row] * expf(Pm[g][row] - M);
    size_t o = (size_t)(rbase + row) * nchunk + blockIdx.y;
    pm_out[o] = M;
    ps_out[o] = S;
  }
}

// -------------------- merge chunk partials -> loss accumulation --------------------

__global__ void flce_merge_kernel(const float* __restrict__ pm,
                                  const float* __restrict__ ps,
                                  const float* __restrict__ tgtl,
                                  const int* __restrict__ target,
                                  float* __restrict__ acc, int N, int nchunk) {
  int row = blockIdx.x * blockDim.x + threadIdx.x;
  if (row >= N) return;
  const float* pmr = pm + (size_t)row * nchunk;
  const float* psr = ps + (size_t)row * nchunk;
  float M = NEG_INF;
  for (int c = 0; c < nchunk; ++c) M = fmaxf(M, pmr[c]);
  float S = 0.0f;
  for (int c = 0; c < nchunk; ++c) S += psr[c] * expf(pmr[c] - M);
  float lse = M + logf(S);
  int t = target[row];
  bool valid = (t != IGNORE_INDEX);
  float z = Z_SCALE * lse * lse;
  float per = valid ? (lse - tgtl[row] + z) : 0.0f;
  float zz = valid ? z : 0.0f;
  atomicAdd(&acc[0], per);
  atomicAdd(&acc[1], zz);
  if (valid) atomicAdd(&acc[2], 1.0f);
}

// -------------------- launch --------------------

extern "C" void kernel_launch(void* const* d_in, const int* in_sizes, int n_in,
                              void* d_out, int out_size, void* d_ws, size_t ws_size,
                              hipStream_t stream) {
  const float* input  = (const float*)d_in[0];
  const float* weight = (const float*)d_in[1];
  const float* bias   = (const float*)d_in[2];
  const int*   target = (const int*)d_in[3];
  float* out = (float*)d_out;

  const int N = in_sizes[3];
  const int V = in_sizes[2];
  const int nchunk = (V + BLK_COLS - 1) / BLK_COLS;

  float* acc  = (float*)d_ws;           // 4 floats
  float* tgtl = acc + 4;                // N floats
  float* pm   = tgtl + N;               // N * nchunk
  float* ps   = pm + (size_t)N * nchunk;

  flce_init_kernel<<<1, 64, 0, stream>>>(acc);

  flce_tgt_kernel<<<(N + 7) / 8, 256, 0, stream>>>(input, weight, bias, target,
                                                   tgtl, N, V);

  // x = row-blocks (fastest): all row-blocks of one 2MB weight chunk run
  // back-to-back so the chunk stays L2-resident; weight crosses HBM ~once.
  dim3 grid(N / BLK_ROWS, nchunk);
  flce_main_kernel<<<grid, NTHREADS, 0, stream>>>(input, weight, bias, pm, ps,
                                                  V, nchunk);

  flce_merge_kernel<<<(N + 255) / 256, 256, 0, stream>>>(pm, ps, tgtl, target,
                                                         acc, N, nchunk);

  flce_finalize_kernel<<<1, 64, 0, stream>>>(acc, out);
}
